// VariationalGCNEncoder_2362232012936
// MI455X (gfx1250) — matrix operand order repaired and checked by reading, not compiled, not run on hardware
//
#include <hip/hip_runtime.h>

// ---------------------------------------------------------------------------
// Variational GCN encoder for MI455X (gfx1250, wave32, WMMA).
// Memory-bound aggregation (atomic scatter) + WMMA bf16-hi/lo-split GEMMs.
// All fp32->bf16 conversion is hoisted into streaming pack kernels; the GEMM
// inner loop is pure (LDS/global b128 loads -> v_wmma_f32_16x16x32_bf16).
// ---------------------------------------------------------------------------

typedef __attribute__((ext_vector_type(16))) __bf16 v16bf;
typedef __attribute__((ext_vector_type(8)))  __bf16 v8bf;
typedef __attribute__((ext_vector_type(8)))  float  v8f;

#define N_NODES 50000
#define N_EDGES 800000
#define DH      128   // hidden / input width (also combined mu|ls width)

// float -> bf16 round-to-nearest-even
static __device__ __forceinline__ __bf16 f2bf(float f) {
    unsigned u = __builtin_bit_cast(unsigned, f);
    unsigned r = (u + 0x7FFFu + ((u >> 16) & 1u)) >> 16;
    unsigned short s = (unsigned short)r;
    return __builtin_bit_cast(__bf16, s);
}
// bf16 -> float (exact)
static __device__ __forceinline__ float bf2f(__bf16 h) {
    unsigned short s = __builtin_bit_cast(unsigned short, h);
    unsigned u = ((unsigned)s) << 16;
    return __builtin_bit_cast(float, u);
}

// ---------------------------------------------------------------------------
// Degree / normalization
// ---------------------------------------------------------------------------
__global__ void k_fill(float* __restrict__ p, float v, int n) {
    int i = blockIdx.x * blockDim.x + threadIdx.x;
    if (i < n) p[i] = v;
}

__global__ void k_count_deg(const long long* __restrict__ dst,
                            float* __restrict__ deg, int e) {
    int i = blockIdx.x * blockDim.x + threadIdx.x;
    if (i < e) atomicAdd(&deg[(int)dst[i]], 1.0f);
}

__global__ void k_dinv(const float* __restrict__ deg,
                       float* __restrict__ dinv, int n) {
    int i = blockIdx.x * blockDim.x + threadIdx.x;
    if (i < n) dinv[i] = rsqrtf(deg[i]);   // deg >= 1 (self loop)
}

// ---------------------------------------------------------------------------
// Weight packing: transposed bf16 hi/lo.
//   W1t*[n][k]  from W1[k][n]            (128x128)
//   W2t*[n][k]  from [W_mu | W_ls][k][n] (128x128; cols 0-63 mu, 64-127 ls)
// Transposed layout makes each lane's B fragment 16 contiguous bf16 (32 B).
// ---------------------------------------------------------------------------
__global__ void k_pack_w(const float* __restrict__ W1,
                         const float* __restrict__ Wmu,
                         const float* __restrict__ Wls,
                         __bf16* __restrict__ W1thi, __bf16* __restrict__ W1tlo,
                         __bf16* __restrict__ W2thi, __bf16* __restrict__ W2tlo) {
    int t = blockIdx.x * blockDim.x + threadIdx.x;   // 0 .. 128*128-1
    if (t >= DH * DH) return;
    int n = t >> 7, k = t & 127;                     // output index [n][k]
    float w1 = W1[k * DH + n];
    __bf16 h1 = f2bf(w1);
    W1thi[t] = h1;
    W1tlo[t] = f2bf(w1 - bf2f(h1));
    float w2 = (n < 64) ? Wmu[k * 64 + n] : Wls[k * 64 + (n - 64)];
    __bf16 h2 = f2bf(w2);
    W2thi[t] = h2;
    W2tlo[t] = f2bf(w2 - bf2f(h2));
}

// x (fp32) -> hi/lo bf16, row-major [N][128]
__global__ void k_pack_x(const float* __restrict__ X,
                         __bf16* __restrict__ Xhi, __bf16* __restrict__ Xlo,
                         int total) {
    int i = blockIdx.x * blockDim.x + threadIdx.x;
    if (i < total) {
        float v = X[i];
        __bf16 hi = f2bf(v);
        Xhi[i] = hi;
        Xlo[i] = f2bf(v - bf2f(hi));
    }
}

// ---------------------------------------------------------------------------
// WMMA GEMM: Y[3125*16 x 128] = unsplit(Xhi+Xlo)[.. x 128] @ Wt^T[128 x 128]
// One block per 16-row M tile; 8 waves, wave w owns N-tile w (16 cols).
// K walked in 4 steps of 32; hi/lo split -> 3 WMMAs per step (12 total).
// ---------------------------------------------------------------------------
__global__ __launch_bounds__(256)
void k_gemm_bf16split(const __bf16* __restrict__ Xhi,
                      const __bf16* __restrict__ Xlo,
                      const __bf16* __restrict__ Wthi,
                      const __bf16* __restrict__ Wtlo,
                      float* __restrict__ Y) {
    __shared__ __align__(16) __bf16 ahi_s[16 * DH];  // 4 KB
    __shared__ __align__(16) __bf16 alo_s[16 * DH];  // 4 KB

    const int mt  = blockIdx.x;
    const int tid = threadIdx.x;
    const size_t base = (size_t)mt * 16 * DH;

    // stage bf16 A tiles: 2048 elems = 256 x uint4 -> one 16B copy per thread
    ((uint4*)ahi_s)[tid] = ((const uint4*)(Xhi + base))[tid];
    ((uint4*)alo_s)[tid] = ((const uint4*)(Xlo + base))[tid];
    __syncthreads();

    const int wave = tid >> 5;          // N tile index 0..7
    const int lane = tid & 31;
    const int m    = lane & 15;         // A row / B column owned by this lane
    const int half = lane >> 4;         // lane group 0/1
    const int ncol = wave * 16 + m;

    const __bf16* wh = Wthi + (size_t)ncol * DH;
    const __bf16* wl = Wtlo + (size_t)ncol * DH;

    v8f c = {};
    #pragma unroll
    for (int kk = 0; kk < 4; ++kk) {
        const int kbase = kk * 32;
        // A fragment (this code's contiguous k order, matched by B):
        //   lanes 0-15: j0-15 -> K kbase+0..15
        //   lanes16-31: j0-15 -> K kbase+16..31
        // -> two aligned contiguous 8-element LDS loads per fragment.
        const int o0 = m * DH + kbase + half * 16;
        const int o1 = o0 + 8;
        v8bf a0h = *(const v8bf*)&ahi_s[o0];
        v8bf a1h = *(const v8bf*)&ahi_s[o1];
        v8bf a0l = *(const v8bf*)&alo_s[o0];
        v8bf a1l = *(const v8bf*)&alo_s[o1];
        v16bf ah = __builtin_shufflevector(a0h, a1h,
                     0,1,2,3,4,5,6,7,8,9,10,11,12,13,14,15);
        v16bf al = __builtin_shufflevector(a0l, a1l,
                     0,1,2,3,4,5,6,7,8,9,10,11,12,13,14,15);
        // B fragment (same contiguous k order as A): lanes 0-15 K=kbase+0..15,
        // lanes 16-31 K=kbase+16..31 -> one contiguous 32B load (transposed W).
        v16bf bh = *(const v16bf*)&wh[kbase + half * 16];
        v16bf bl = *(const v16bf*)&wl[kbase + half * 16];

        c = __builtin_amdgcn_wmma_f32_16x16x32_bf16(false, ah, false, bh,
                                                    (short)0, c, false, false);
        c = __builtin_amdgcn_wmma_f32_16x16x32_bf16(false, al, false, bh,
                                                    (short)0, c, false, false);
        c = __builtin_amdgcn_wmma_f32_16x16x32_bf16(false, ah, false, bl,
                                                    (short)0, c, false, false);
    }

    // C/D layout: VGPR r -> M = half*8 + r, column = lane&15 within N tile
    float* yrow = Y + base;
    #pragma unroll
    for (int r = 0; r < 8; ++r) {
        int mm = half * 8 + r;
        yrow[mm * DH + ncol] = c[r];
    }
}

// ---------------------------------------------------------------------------
// Aggregation: agg = self-loop init + atomic edge scatter
// ---------------------------------------------------------------------------
__global__ void k_self_init(const float* __restrict__ lin,
                            const float* __restrict__ dinv,
                            float* __restrict__ agg, int total) {
    int i = blockIdx.x * blockDim.x + threadIdx.x;
    if (i < total) {
        float dv = dinv[i >> 7];        // norm of self loop = dinv^2
        agg[i] = dv * dv * lin[i];
    }
}

__global__ __launch_bounds__(256)
void k_edge_scatter(const long long* __restrict__ src,
                    const long long* __restrict__ dst,
                    const float* __restrict__ dinv,
                    const float* __restrict__ lin,
                    float* __restrict__ agg, int e) {
    int gwave = (int)((blockIdx.x * blockDim.x + threadIdx.x) >> 5);
    int lane  = threadIdx.x & 31;
    if (gwave >= e) return;
    int s = (int)src[gwave];
    int d = (int)dst[gwave];
    float coef = dinv[s] * dinv[d];
    const float* ls = lin + (size_t)s * DH;
    float*       ad = agg + (size_t)d * DH;
    #pragma unroll
    for (int cidx = 0; cidx < 4; ++cidx) {
        int f = lane + cidx * 32;       // coalesced across the wave
        atomicAdd(&ad[f], coef * ls[f]);
    }
}

// h = relu(agg + b1), emitted directly as bf16 hi/lo packs for the next GEMM
__global__ void k_relu_bias_pack(const float* __restrict__ agg,
                                 const float* __restrict__ b1,
                                 __bf16* __restrict__ Hhi,
                                 __bf16* __restrict__ Hlo, int total) {
    int i = blockIdx.x * blockDim.x + threadIdx.x;
    if (i < total) {
        float v = agg[i] + b1[i & 127];
        v = v > 0.0f ? v : 0.0f;
        __bf16 hi = f2bf(v);
        Hhi[i] = hi;
        Hlo[i] = f2bf(v - bf2f(hi));
    }
}

// split agg2[:, :64]+b_mu -> out[0:N*64), agg2[:, 64:]+b_ls -> out[N*64:2*N*64)
__global__ void k_final(const float* __restrict__ agg2,
                        const float* __restrict__ bmu,
                        const float* __restrict__ bls,
                        float* __restrict__ out, int n) {
    int i = blockIdx.x * blockDim.x + threadIdx.x;   // 0 .. n*64-1
    if (i < n * 64) {
        int node = i >> 6, f = i & 63;
        out[i]                  = agg2[node * DH + f]      + bmu[f];
        out[(size_t)n * 64 + i] = agg2[node * DH + 64 + f] + bls[f];
    }
}

// ---------------------------------------------------------------------------
// Launch
// ---------------------------------------------------------------------------
extern "C" void kernel_launch(void* const* d_in, const int* in_sizes, int n_in,
                              void* d_out, int out_size, void* d_ws, size_t ws_size,
                              hipStream_t stream) {
    (void)in_sizes; (void)n_in; (void)out_size; (void)ws_size;

    const float*     x   = (const float*)d_in[0];
    const long long* ei  = (const long long*)d_in[1];   // int64 [2, E]
    const float*     W1  = (const float*)d_in[2];
    const float*     b1  = (const float*)d_in[3];
    const float*     Wmu = (const float*)d_in[4];
    const float*     bmu = (const float*)d_in[5];
    const float*     Wls = (const float*)d_in[6];
    const float*     bls = (const float*)d_in[7];
    float*           out = (float*)d_out;

    const long long* srcv = ei;
    const long long* dstv = ei + N_EDGES;

    // workspace carve-up (256B aligned)
    char*  ws  = (char*)d_ws;
    size_t off = 0;
    auto carve = [&](size_t bytes) {
        void* p = ws + off;
        off = (off + bytes + 255) & ~(size_t)255;
        return p;
    };
    float*  deg   = (float*)carve((size_t)N_NODES * 4);
    float*  dinv  = (float*)carve((size_t)N_NODES * 4);
    __bf16* W1thi = (__bf16*)carve((size_t)DH * DH * 2);
    __bf16* W1tlo = (__bf16*)carve((size_t)DH * DH * 2);
    __bf16* W2thi = (__bf16*)carve((size_t)DH * DH * 2);
    __bf16* W2tlo = (__bf16*)carve((size_t)DH * DH * 2);
    __bf16* packhi = (__bf16*)carve((size_t)N_NODES * DH * 2); // x-pack, then h-pack
    __bf16* packlo = (__bf16*)carve((size_t)N_NODES * DH * 2);
    float*  lin   = (float*)carve((size_t)N_NODES * DH * 4);   // reused both layers
    float*  agg   = (float*)carve((size_t)N_NODES * DH * 4);   // reused both layers

    const int total = N_NODES * DH;           // 6.4M
    const int tb = 256;
    const int gN   = (N_NODES + tb - 1) / tb;
    const int gE   = (N_EDGES + tb - 1) / tb;
    const int gTot = (total + tb - 1) / tb;
    const int gEw  = (N_EDGES + 7) / 8;       // 8 waves per block, wave per edge
    const int gMt  = N_NODES / 16;            // 3125 (exact)
    const int gOut = (N_NODES * 64 + tb - 1) / tb;

    // normalization
    k_fill<<<gN, tb, 0, stream>>>(deg, 1.0f, N_NODES);
    k_count_deg<<<gE, tb, 0, stream>>>(dstv, deg, N_EDGES);
    k_dinv<<<gN, tb, 0, stream>>>(deg, dinv, N_NODES);

    // packs
    k_pack_w<<<(DH * DH) / tb, tb, 0, stream>>>(W1, Wmu, Wls,
                                                W1thi, W1tlo, W2thi, W2tlo);
    k_pack_x<<<gTot, tb, 0, stream>>>(x, packhi, packlo, total);

    // layer 1: lin = x@W1 ; agg = scatter(norm*lin) ; h-pack = relu(agg + b1)
    k_gemm_bf16split<<<gMt, tb, 0, stream>>>(packhi, packlo, W1thi, W1tlo, lin);
    k_self_init<<<gTot, tb, 0, stream>>>(lin, dinv, agg, total);
    k_edge_scatter<<<gEw, tb, 0, stream>>>(srcv, dstv, dinv, lin, agg, N_EDGES);
    k_relu_bias_pack<<<gTot, tb, 0, stream>>>(agg, b1, packhi, packlo, total);

    // layer 2 (mu | logstd fused): lin = h@[Wmu|Wls] ; agg = scatter ; out
    k_gemm_bf16split<<<gMt, tb, 0, stream>>>(packhi, packlo, W2thi, W2tlo, lin);
    k_self_init<<<gTot, tb, 0, stream>>>(lin, dinv, agg, total);
    k_edge_scatter<<<gEw, tb, 0, stream>>>(srcv, dstv, dinv, lin, agg, N_EDGES);
    k_final<<<gOut, tb, 0, stream>>>(agg, bmu, bls, out, N_NODES);
}
